// MxCorrelation_41042707480848
// MI455X (gfx1250) — matrix operand order repaired and checked by reading, not compiled, not run on hardware
//
#include <hip/hip_runtime.h>
#include <hip/hip_bf16.h>

// FlowNetC correlation on MI455X (gfx1250), K=1/S1=1/S2=2/MD=PAD=20 specialization:
// out[b, dyi*21+dxi, y, x] = (1/128) * sum_c in1[b,c,y,x]*in2[b,c,y+2(dyi-10),x+2(dxi-10)]
// Strategy: parity-split banded GEMM on v_wmma_f32_16x16x32_f16 with LDS staging,
// float4 staging loads, branch-free LDS scatter, non-temporal float4 output stores.

typedef _Float16 v16h __attribute__((ext_vector_type(16)));
typedef _Float16 v8h  __attribute__((ext_vector_type(8)));
typedef float    v8f  __attribute__((ext_vector_type(8)));
typedef float    v4f  __attribute__((ext_vector_type(4)));

#define LDR 136  // padded LDS row stride in halfs (128 ch + 8 pad) => 272B, 16B-aligned rows

__global__ __launch_bounds__(288) void MxCorrelation_kernel(
    const float* __restrict__ in1, const float* __restrict__ in2,
    float* __restrict__ out)
{
  constexpr int C = 128, H = 96, W = 96, ND = 21;

  // A: input1 row, parity-separated: [parity][p=0..47][c]
  __shared__ __align__(16) _Float16 sA[2 * 48 * LDR];
  // B: input2 row with +-16 parity-position halo: [parity][qq=0..79][c], q = qq-16
  __shared__ __align__(16) _Float16 sB[2 * 80 * LDR];
  // Output tile [dxi=0..20][x=0..95], stride 97 vs bank conflicts; row 21 = trash row
  __shared__ float sO[22 * 97];

  const int tid = threadIdx.x;
  const int y   = blockIdx.x;   // output row
  const int dyi = blockIdx.y;   // vertical displacement index
  const int b   = blockIdx.z;   // batch
  const int y2  = y + (dyi - 10) * 2;

  // ---- Stage input1 row y -> sA (float4 loads, fp32 -> f16) ----
  {
    const size_t base1 = (size_t)b * C * H * W + (size_t)y * W;
    for (int j = tid; j < C * (W / 4); j += 288) {
      int c = j / (W / 4), xq = j - c * (W / 4);
      v4f v = *(const v4f*)(in1 + base1 + (size_t)c * H * W + 4 * xq);
      int p = 2 * xq;  // x = 4*xq -> elements at parities 0,1,0,1 and p, p, p+1, p+1
      sA[(0 * 48 + p    ) * LDR + c] = (_Float16)v.x;
      sA[(1 * 48 + p    ) * LDR + c] = (_Float16)v.y;
      sA[(0 * 48 + p + 1) * LDR + c] = (_Float16)v.z;
      sA[(1 * 48 + p + 1) * LDR + c] = (_Float16)v.w;
    }
  }

  // ---- Stage input2 row y2 (halo, zero-padded) -> sB ----
  // x2 = 4*k4 - 32, step 4: every float4 slot is fully in [0,96) or fully OOB.
  {
    const bool rowok = (unsigned)y2 < (unsigned)H;
    const size_t base2 = (size_t)b * C * H * W + (size_t)y2 * W;
    for (int j = tid; j < C * 40; j += 288) {
      int c = j / 40, k4 = j - c * 40;
      int x2 = 4 * k4 - 32;
      v4f v = {0.f, 0.f, 0.f, 0.f};
      if (rowok && (unsigned)x2 < (unsigned)W)
        v = *(const v4f*)(in2 + base2 + (size_t)c * H * W + x2);
      int qq = 2 * k4;  // qq = (x2+32-parity)/2
      sB[(0 * 80 + qq    ) * LDR + c] = (_Float16)v.x;
      sB[(1 * 80 + qq    ) * LDR + c] = (_Float16)v.y;
      sB[(0 * 80 + qq + 1) * LDR + c] = (_Float16)v.z;
      sB[(1 * 80 + qq + 1) * LDR + c] = (_Float16)v.w;
    }
  }
  __syncthreads();

  // ---- Banded GEMM: 18 macro-tiles over 9 waves, 4 chained WMMAs each (K=128) ----
  const int wave = tid >> 5;
  const int lane = tid & 31;
  const int ln16 = lane & 15;     // M index for A row / N index for B column
  const int hi   = lane >> 4;
  const int kbA  = hi ? 8 : 0;    // A: lanes 16-31 hold K = 8..15 / 24..31
  const int khB  = hi ? 16 : 0;   // unused: B loads use kbA, so B lanes 16-31 hold K = 8..15 / 24..31 like A

  for (int t = wave; t < 18; t += 9) {
    const int parity = t / 9;
    const int rem    = t - parity * 9;
    const int p0     = (rem / 3) * 16;        // input1 parity-pixel tile base
    const int qoff   = (rem % 3) * 16 - 16;   // diagonal band offset {-16, 0, 16}

    const _Float16* pa = &sA[(parity * 48 + p0 + ln16) * LDR];
    const _Float16* pb = &sB[(parity * 80 + (p0 + qoff + 16) + ln16) * LDR];

    v8f acc = {};
#pragma unroll
    for (int ks = 0; ks < 4; ++ks) {
      v8h alo = *(const v8h*)(pa + ks * 32 + kbA);
      v8h ahi = *(const v8h*)(pa + ks * 32 + 16 + kbA);
      v16h a = __builtin_shufflevector(alo, ahi, 0,1,2,3,4,5,6,7,8,9,10,11,12,13,14,15);
      v8h blo = *(const v8h*)(pb + ks * 32 + kbA);
      v8h bhi = *(const v8h*)(pb + ks * 32 + 16 + kbA);
      v16h bm = __builtin_shufflevector(blo, bhi, 0,1,2,3,4,5,6,7,8,9,10,11,12,13,14,15);
      acc = __builtin_amdgcn_wmma_f32_16x16x32_f16(
          /*neg_a=*/false, a, /*neg_b=*/false, bm,
          /*c_mod=*/(short)0, acc, /*reuse_a=*/false, /*reuse_b=*/false);
    }

    // D layout: VGPR r -> M = r + 8*hi, N = lane&15. d2 = qoff + n - m valid in [-10,10].
    // Branch-free scatter: out-of-band elements go to per-lane slots in trash row 21.
#pragma unroll
    for (int r = 0; r < 8; ++r) {
      int mm = r + 8 * hi;
      int d2 = qoff + ln16 - mm;
      unsigned dxi = (unsigned)(d2 + 10);
      int x   = 2 * (p0 + mm) + parity;
      int idx = (dxi < (unsigned)ND) ? (int)dxi * 97 + x : ND * 97 + lane;
      sO[idx] = acc[r];
    }
  }
  __syncthreads();

  // ---- Non-temporal coalesced store: 21 rows of 96 floats, scaled by 1/C ----
  // Output is write-once/never-reread; NT keeps the L2-resident inputs unevicted.
  {
    const size_t obase =
        ((size_t)b * ND * ND + (size_t)dyi * ND) * (size_t)(H * W) + (size_t)y * W;
    for (int j = tid; j < ND * (W / 4); j += 288) {
      int dxi = j / (W / 4), xq = j - dxi * (W / 4);
      int x = 4 * xq;
      const float* s = &sO[dxi * 97 + x];
      v4f o;
      o.x = s[0] * (1.0f / 128.0f);
      o.y = s[1] * (1.0f / 128.0f);
      o.z = s[2] * (1.0f / 128.0f);
      o.w = s[3] * (1.0f / 128.0f);
      __builtin_nontemporal_store(o, (v4f*)(out + obase + (size_t)dxi * H * W + x));
    }
  }
}

extern "C" void kernel_launch(void* const* d_in, const int* in_sizes, int n_in,
                              void* d_out, int out_size, void* d_ws, size_t ws_size,
                              hipStream_t stream) {
  const float* in1 = (const float*)d_in[0];
  const float* in2 = (const float*)d_in[1];
  float* out = (float*)d_out;
  dim3 grid(96, 21, 4);   // (y, dyi, b)
  dim3 block(288);        // 9 waves of 32
  hipLaunchKernelGGL(MxCorrelation_kernel, grid, block, 0, stream, in1, in2, out);
}
